// XNOR_case1_BinLinear_28200755265772
// MI455X (gfx1250) — 8-bit matrix operand order repaired and checked by reading, compiled, not run on hardware
//
#include <hip/hip_runtime.h>

#define N_ROWS 262144
#define K_IN   256
#define M_OUT  256

typedef __attribute__((ext_vector_type(16))) int   v16i;
typedef __attribute__((ext_vector_type(8)))  float v8f;

__device__ __forceinline__ unsigned int pack4_fp8(uint4 v) {
  unsigned int p1  = __builtin_amdgcn_perm(v.y, v.x, 0x07030703u);
  unsigned int p2  = __builtin_amdgcn_perm(v.w, v.z, 0x07030703u);
  unsigned int msb = __builtin_amdgcn_perm(p2, p1, 0x05040100u);
  return (msb & 0x80808080u) | 0x38383838u;
}

__global__ __launch_bounds__(256)
void xnor_binlinear_wmma(const float* __restrict__ x,
                         const float* __restrict__ w,
                         const float* __restrict__ scale,
                         float* __restrict__ y) {
  __shared__ uint4 ldsw[M_OUT * 16];

  const int tid = threadIdx.x;

  {
    const int row = tid;
    const uint4* wrow = (const uint4*)(w + (size_t)row * K_IN);
    #pragma unroll 4
    for (int gq = 0; gq < 16; ++gq) {
      uint4 f0 = wrow[gq * 4 + 0];
      uint4 f1 = wrow[gq * 4 + 1];
      uint4 f2 = wrow[gq * 4 + 2];
      uint4 f3 = wrow[gq * 4 + 3];
      uint4 g;
      g.x = pack4_fp8(f0);
      g.y = pack4_fp8(f1);
      g.z = pack4_fp8(f2);
      g.w = pack4_fp8(f3);
      ldsw[row * 16 + (gq ^ (row & 15))] = g;
    }
  }
  __syncthreads();

  const int wave = tid >> 5;
  const int lane = tid & 31;
  const int g    = lane >> 4;
  const int c    = lane & 15;
  const int mbase = (blockIdx.x * 8 + wave) * 16;

  v16i a0, a1;
  const float* xrow = x + (size_t)(mbase + c) * K_IN;
  #pragma unroll
  for (int j = 0; j < 16; ++j) {
    const int jj  = j & 7;
    const int mat = j >> 3;
    const int k   = ((j >> 2) << 5) + (g << 4) + ((j & 3) << 2);
    a0[j] = (int)pack4_fp8(*(const uint4*)(xrow + k));
    a1[j] = (int)pack4_fp8(*(const uint4*)(xrow + 128 + k));
  }

  #pragma unroll 2
  for (int ot = 0; ot < 16; ++ot) {
    const int o = ot * 16 + c;

    v16i b0, b1;
    #pragma unroll
    for (int blk = 0; blk < 4; ++blk) {
      const int q0 = (blk * 2 + g);
      const int q1 = (8 + blk * 2 + g);
      uint4 u0 = ldsw[o * 16 + (q0 ^ (o & 15))];
      uint4 u1 = ldsw[o * 16 + (q1 ^ (o & 15))];
      b0[4 * blk + 0] = (int)u0.x; b0[4 * blk + 1] = (int)u0.y;
      b0[4 * blk + 2] = (int)u0.z; b0[4 * blk + 3] = (int)u0.w;
      b1[4 * blk + 0] = (int)u1.x; b1[4 * blk + 1] = (int)u1.y;
      b1[4 * blk + 2] = (int)u1.z; b1[4 * blk + 3] = (int)u1.w;
    }

    v8f acc = {};
    acc = __builtin_amdgcn_wmma_f32_16x16x128_fp8_fp8(a0, b0, (short)0, acc,
                                                      false, false);
    acc = __builtin_amdgcn_wmma_f32_16x16x128_fp8_fp8(a1, b1, (short)0, acc,
                                                      false, false);

    const float s = scale[o];
    #pragma unroll
    for (int v = 0; v < 8; ++v) {
      y[(size_t)(mbase + v + 8 * g) * M_OUT + o] = acc[v] * s;
    }
  }
}

extern "C" void kernel_launch(void* const* d_in, const int* in_sizes, int n_in,
                              void* d_out, int out_size, void* d_ws, size_t ws_size,
                              hipStream_t stream) {
  const float* x     = (const float*)d_in[0];
  const float* w     = (const float*)d_in[1];
  const float* scale = (const float*)d_in[2];
  float*       y     = (float*)d_out;

  dim3 grid(N_ROWS / (16 * 8));
  dim3 block(256);
  xnor_binlinear_wmma<<<grid, block, 0, stream>>>(x, w, scale, y);
}
